// GNN_54211077210202
// MI455X (gfx1250) — matrix operand order repaired and checked by reading, not compiled, not run on hardware
//
#include <hip/hip_runtime.h>

// ---------- types ----------
typedef __bf16 bf16;
typedef __attribute__((ext_vector_type(4)))  float          f32x4;
typedef __attribute__((ext_vector_type(4)))  unsigned short u16x4;
typedef __attribute__((ext_vector_type(8)))  bf16           v8bf;
typedef __attribute__((ext_vector_type(16))) bf16           v16bf;
typedef __attribute__((ext_vector_type(8)))  float          v8f;
typedef __attribute__((ext_vector_type(4)))  unsigned int   u32x4;
typedef __attribute__((ext_vector_type(8)))  int            i32x8;
typedef __attribute__((ext_vector_type(4)))  int            i32x4;

// f32 -> bf16, round to nearest even
__device__ __forceinline__ bf16 f2bf(float f) {
    unsigned u = __builtin_bit_cast(unsigned, f);
    u += 0x7FFFu + ((u >> 16) & 1u);
    return __builtin_bit_cast(bf16, (unsigned short)(u >> 16));
}

__device__ __forceinline__ v16bf cat8(v8bf a, v8bf b) {
    return __builtin_shufflevector(a, b, 0,1,2,3,4,5,6,7,8,9,10,11,12,13,14,15);
}

// ---------- node embedding: h[n] = x_emb1[atom] + x_emb2[chi] ----------
__global__ __launch_bounds__(256) void node_embed(
    const int* __restrict__ x, const float* __restrict__ xe1,
    const float* __restrict__ xe2, float* __restrict__ h, int N)
{
    int gid = blockIdx.x * blockDim.x + threadIdx.x;
    int n = gid >> 5;
    if (n >= N) return;
    int d = (gid & 31) * 4;
    int a = x[n * 2 + 0], c = x[n * 2 + 1];
    f32x4 v1 = *(const f32x4*)(xe1 + (size_t)a * 128 + d);
    f32x4 v2 = *(const f32x4*)(xe2 + (size_t)c * 128 + d);
    *(f32x4*)(h + (size_t)n * 128 + d) = v1 + v2;
}

// ---------- agg init with self-loop: agg[n] = h[n] + se1 + se2 + se3 ----------
__global__ __launch_bounds__(256) void agg_init(
    const float* __restrict__ h, const float* __restrict__ se1,
    const float* __restrict__ se2, const float* __restrict__ se3,
    float* __restrict__ agg, int N)
{
    int gid = blockIdx.x * blockDim.x + threadIdx.x;
    int n = gid >> 5;
    if (n >= N) return;
    int d = (gid & 31) * 4;
    f32x4 hv = *(const f32x4*)(h + (size_t)n * 128 + d);
    f32x4 e1 = *(const f32x4*)(se1 + d);
    f32x4 e2 = *(const f32x4*)(se2 + d);
    f32x4 e3 = *(const f32x4*)(se3 + d);
    *(f32x4*)(agg + (size_t)n * 128 + d) = hv + e1 + e2 + e3;
}

// ---------- edge scatter: agg[col] += h[row] + edge_emb  (wave per edge) ----------
__global__ __launch_bounds__(256) void edge_scatter(
    const float* __restrict__ h, const int* __restrict__ eidx,
    const int* __restrict__ eattr,
    const float* __restrict__ e1, const float* __restrict__ e2,
    const float* __restrict__ e3, float* __restrict__ agg, int E)
{
    int gid = blockIdx.x * blockDim.x + threadIdx.x;
    int e = gid >> 5;
    if (e >= E) return;
    int lane = gid & 31;
    int d = lane * 4;
    int r  = eidx[e];
    int c  = eidx[E + e];
    int a0 = eattr[e * 3 + 0];
    int a1 = eattr[e * 3 + 1];
    int a2 = eattr[e * 3 + 2];
    f32x4 hv  = *(const f32x4*)(h  + (size_t)r  * 128 + d);
    f32x4 v1  = *(const f32x4*)(e1 + (size_t)a0 * 128 + d);
    f32x4 v2  = *(const f32x4*)(e2 + (size_t)a1 * 128 + d);
    f32x4 v3  = *(const f32x4*)(e3 + (size_t)a2 * 128 + d);
    f32x4 m = hv + v1 + v2 + v3;
    float* dst = agg + (size_t)c * 128 + d;
    __hip_atomic_fetch_add(dst + 0, m.x, __ATOMIC_RELAXED, __HIP_MEMORY_SCOPE_AGENT);
    __hip_atomic_fetch_add(dst + 1, m.y, __ATOMIC_RELAXED, __HIP_MEMORY_SCOPE_AGENT);
    __hip_atomic_fetch_add(dst + 2, m.z, __ATOMIC_RELAXED, __HIP_MEMORY_SCOPE_AGENT);
    __hip_atomic_fetch_add(dst + 3, m.w, __ATOMIC_RELAXED, __HIP_MEMORY_SCOPE_AGENT);
}

// ---------- GEMM helpers ----------
__device__ __forceinline__ void loadA4(const float* g, bf16* d) {
    f32x4 v = *(const f32x4*)g;
    d[0] = f2bf(v.x); d[1] = f2bf(v.y); d[2] = f2bf(v.z); d[3] = f2bf(v.w);
}
__device__ __forceinline__ void storeOut(float* p, float v) { *p = v; }
__device__ __forceinline__ void storeOut(unsigned short* p, float v) {
    *p = __builtin_bit_cast(unsigned short, f2bf(v));
}

// ---------- tiled WMMA GEMM:  C[M,NNT] = act(A[M,K] @ B[K,NNT] + bias) ----------
// Block covers the FULL N dimension (NNT = 256 or 128) so A is read exactly once.
// USE_TDM: stage the (already-bf16) A tile with the Tensor Data Mover, using
// hardware LDS padding (16 DW interval, 4 DW pad) to reproduce the LDT=40 layout.
#define BM 128
#define BK 32
#define LDT 40   // BK + 8 pad (bf16 units)

template <typename AT, typename OT, int NNT, bool USE_TDM>
__global__ __launch_bounds__(256) void gemm_wmma(
    const AT* __restrict__ A, const float* __restrict__ B,
    const float* __restrict__ bias, OT* __restrict__ C,
    int M, int K, int doRelu)
{
    constexpr int NS = NNT / 32;          // n-subtiles (16 wide) per wave
    __shared__ bf16 sA[BM * LDT];         // 10 KB
    __shared__ bf16 sB[NNT * LDT];        // 20 KB (NNT=256) / 10 KB

    const int tid  = threadIdx.x;
    const int lane = tid & 31;
    const int wave = tid >> 5;
    const int wm   = wave >> 1;           // 0..3 -> 32-row sub-block
    const int wn   = wave & 1;            // 0..1 -> NNT/2-col sub-block
    const int ln   = lane & 15;
    const int hsel = lane >> 4;           // half-wave select
    const int rowBase = blockIdx.x * BM;
    const bf16 BZ = __builtin_bit_cast(bf16, (unsigned short)0);

    v8f acc[2][NS] = {};

    for (int kb = 0; kb < K; kb += BK) {
        // --- stage A tile (BM x BK) ---
        if constexpr (USE_TDM) {
            if (wave == 0) {
                // Tensor DMA descriptor (CDNA5 ISA ch.8): 2D bf16 tile
                unsigned ldsA = (unsigned)(unsigned long long)(&sA[0]); // flat[31:0] == LDS offset
                unsigned long long ga =
                    (unsigned long long)(const void*)(A + (size_t)rowBase * K + kb);
                int remRows = M - rowBase;           // TDM zero-fills rows beyond this
                u32x4 g0;
                g0[0] = 1u;                                          // count=1, no gather
                g0[1] = ldsA;                                        // lds_addr
                g0[2] = (unsigned)(ga & 0xFFFFFFFFull);              // global_addr lo
                g0[3] = (unsigned)((ga >> 32) & 0x01FFFFFFull)       // global_addr hi
                        | (2u << 30);                                // type=2 (image)
                i32x8 g1;
                g1[0] = (1 << 16)        // data_size = 2 bytes
                      | (1 << 20)        // pad_enable
                      | (3 << 22)        // pad_interval: 16 DWORDs (one 64B tile row)
                      | (3 << 25);       // pad_amount:   4 DWORDs (16B -> LDT=40)
                g1[1] = (K & 0xFFFF) << 16;                          // tensor_dim0[15:0]
                g1[2] = ((K >> 16) & 0xFFFF)                         // tensor_dim0[31:16]
                      | ((remRows & 0xFFFF) << 16);                  // tensor_dim1[15:0]
                g1[3] = ((remRows >> 16) & 0xFFFF)                   // tensor_dim1[31:16]
                      | (BK << 16);                                  // tile_dim0 = 32
                g1[4] = BM;                                          // tile_dim1 = 128
                g1[5] = K;                                           // dim0_stride lo
                g1[6] = 0;                                           // dim0_stride hi
                g1[7] = 0;
                i32x4 z4 = {0, 0, 0, 0};
                i32x8 z8 = {0, 0, 0, 0, 0, 0, 0, 0};
                __builtin_amdgcn_tensor_load_to_lds(g0, g1, z4, z4, z8, 0);
                __builtin_amdgcn_s_wait_tensorcnt(0);
            }
        } else {
            int c4 = (tid & 7) * 4;       // 0..28
            int r0 = tid >> 3;            // 0..31
#pragma unroll
            for (int i = 0; i < 4; ++i) {
                int r  = r0 + i * 32;
                int gr = rowBase + r;
                bf16* dst = &sA[r * LDT + c4];
                if (gr < M) {
                    loadA4((const float*)A + (size_t)gr * K + kb + c4, dst);
                } else {
                    dst[0] = BZ; dst[1] = BZ; dst[2] = BZ; dst[3] = BZ;
                }
            }
        }
        // --- stage B tile transposed: sB[col][k]  (BK x NNT source, row-major) ---
        {
            constexpr int TPR  = NNT / 4;      // threads per B row (float4 granules)
            constexpr int KSTR = 256 / TPR;    // k-rows covered per pass
            int c4  = (tid % TPR) * 4;
            int kr0 = tid / TPR;
#pragma unroll
            for (int i = 0; i < BK / KSTR; ++i) {
                int kr = kr0 + i * KSTR;
                f32x4 v = *(const f32x4*)&B[(size_t)(kb + kr) * NNT + c4];
                sB[(c4 + 0) * LDT + kr] = f2bf(v.x);
                sB[(c4 + 1) * LDT + kr] = f2bf(v.y);
                sB[(c4 + 2) * LDT + kr] = f2bf(v.z);
                sB[(c4 + 3) * LDT + kr] = f2bf(v.w);
            }
        }
        __syncthreads();

        // --- A fragments per ISA 16-bit layout ---
        // lanes 0-15 hold K{0..7,16..23}, lanes 16-31 hold K{8..15,24..31}
        v16bf aF[2];
        const int aK = hsel ? 8 : 0;
#pragma unroll
        for (int s = 0; s < 2; ++s) {
            const bf16* p = &sA[(wm * 32 + s * 16 + ln) * LDT + aK];
            v8bf lo = *(const v8bf*)p;
            v8bf hi = *(const v8bf*)(p + 16);
            aF[s] = cat8(lo, hi);
        }
        // --- stream n-subtiles, software-pipelining the B fragment loads ---
        const int bK = hsel ? 8 : 0;
        const int bColBase = wn * (NNT / 2) + ln;
        {
            const bf16* p = &sB[bColBase * LDT + bK];
            v8bf lo = *(const v8bf*)p;
            v8bf hi = *(const v8bf*)(p + 16);
            v16bf bF = cat8(lo, hi);
#pragma unroll
            for (int ni = 0; ni < NS; ++ni) {
                v16bf bCur = bF;
                if (ni + 1 < NS) {
                    const bf16* q = &sB[(bColBase + (ni + 1) * 16) * LDT + bK];
                    v8bf qlo = *(const v8bf*)q;
                    v8bf qhi = *(const v8bf*)(q + 16);
                    bF = cat8(qlo, qhi);
                }
                acc[0][ni] = __builtin_amdgcn_wmma_f32_16x16x32_bf16(
                    false, aF[0], false, bCur, (short)0, acc[0][ni], false, false);
                acc[1][ni] = __builtin_amdgcn_wmma_f32_16x16x32_bf16(
                    false, aF[1], false, bCur, (short)0, acc[1][ni], false, false);
            }
        }
        __syncthreads();
    }

    // --- epilogue: bias + activation ---
#pragma unroll
    for (int mi = 0; mi < 2; ++mi) {
#pragma unroll
        for (int ni = 0; ni < NS; ++ni) {
            int col = wn * (NNT / 2) + ni * 16 + ln;
            float bv = bias[col];
#pragma unroll
            for (int r = 0; r < 8; ++r) {
                int row = rowBase + wm * 32 + mi * 16 + hsel * 8 + r;
                if (row < M) {
                    float v = acc[mi][ni][r] + bv;
                    if (doRelu) v = fmaxf(v, 0.0f);
                    storeOut(&C[(size_t)row * NNT + col], v);
                }
            }
        }
    }
}

// ---------- host launcher ----------
extern "C" void kernel_launch(void* const* d_in, const int* in_sizes, int n_in,
                              void* d_out, int out_size, void* d_ws, size_t ws_size,
                              hipStream_t stream)
{
    const int*   x     = (const int*)d_in[0];
    const int*   eidx  = (const int*)d_in[1];
    const int*   eattr = (const int*)d_in[2];
    const float* xe1   = (const float*)d_in[3];
    const float* xe2   = (const float*)d_in[4];
    const float* ee1   = (const float*)d_in[5];   // (L, 7, D)
    const float* ee2   = (const float*)d_in[6];   // (L, 3, D)
    const float* ee3   = (const float*)d_in[7];   // (L, 3, D)
    const float* W1    = (const float*)d_in[8];   // (L, D, 2D)
    const float* b1    = (const float*)d_in[9];   // (L, 2D)
    const float* W2    = (const float*)d_in[10];  // (L, 2D, D)
    const float* b2    = (const float*)d_in[11];  // (L, D)

    const int N = in_sizes[0] / 2;
    const int E = in_sizes[1] / 2;
    const int D = 128, L = 5;

    float* h   = (float*)d_ws;                                    // N*D f32
    float* agg = h + (size_t)N * D;                               // N*D f32
    unsigned short* tmp = (unsigned short*)(agg + (size_t)N * D); // N*2D bf16

    dim3 blk(256);
    int nodeThreads = N * 32;
    int edgeThreads = E * 32;

    node_embed<<<(nodeThreads + 255) / 256, blk, 0, stream>>>(x, xe1, xe2, h, N);

    for (int l = 0; l < L; ++l) {
        const float* e1l = ee1 + (size_t)l * 7 * D;
        const float* e2l = ee2 + (size_t)l * 3 * D;
        const float* e3l = ee3 + (size_t)l * 3 * D;

        // self-loop attr = (4, 0, 0)
        agg_init<<<(nodeThreads + 255) / 256, blk, 0, stream>>>(
            h, e1l + 4 * D, e2l, e3l, agg, N);
        edge_scatter<<<(edgeThreads + 255) / 256, blk, 0, stream>>>(
            h, eidx, eattr, e1l, e2l, e3l, agg, E);

        // GEMM1: relu(agg @ W1 + b1) -> tmp (bf16), full-N block (NNT=256)
        dim3 g1((N + BM - 1) / BM);
        gemm_wmma<float, unsigned short, 256, false><<<g1, blk, 0, stream>>>(
            agg, W1 + (size_t)l * D * 2 * D, b1 + (size_t)l * 2 * D,
            tmp, N, D, 1);

        // GEMM2: tmp @ W2 + b2 -> h (f32); last layer -> d_out, no relu (NNT=128)
        // A tile (bf16) staged via Tensor Data Mover.
        float* outp = (l == L - 1) ? (float*)d_out : h;
        gemm_wmma<unsigned short, float, 128, true><<<g1, blk, 0, stream>>>(
            tmp, W2 + (size_t)l * 2 * D * D, b2 + (size_t)l * D,
            outp, N, 2 * D, (l < L - 1) ? 1 : 0);
    }
}
